// FLK_NKRR_10393820856369
// MI455X (gfx1250) — matrix operand order repaired and checked by reading, not compiled, not run on hardware
//
#include <hip/hip_runtime.h>
#include <hip/hip_bf16.h>

typedef __attribute__((ext_vector_type(16))) __bf16 v16bf;
typedef __attribute__((ext_vector_type(8)))  float  v8f;

#define DD 128            // feature dim (fixed by problem)
#define LOG2E 1.4426950408889634f

union BU { uint4 q[2]; v16bf v; };

__device__ __forceinline__ void cvt_split(float x, v16bf& h, v16bf& l, int e) {
    __bf16 hh = (__bf16)x;
    h[e] = hh;
    l[e] = (__bf16)(x - (float)hh);
}

// ---------------------------------------------------------------------------
// Prep: centers (fp32 [M,128]) -> Chi/Clo (bf16 split), cn[m] = ||c_m||^2.
// Block 0 thread 0 also zeroes the two scalar accumulators.
// ---------------------------------------------------------------------------
__global__ __launch_bounds__(128) void flk_prep(const float* __restrict__ centers,
                                                __bf16* __restrict__ Chi,
                                                __bf16* __restrict__ Clo,
                                                float* __restrict__ cn,
                                                float* __restrict__ scal) {
    __shared__ float red[128];
    const int m = blockIdx.x, t = threadIdx.x;
    const float x = centers[(size_t)m * DD + t];
    __bf16 h = (__bf16)x;
    Chi[(size_t)m * DD + t] = h;
    Clo[(size_t)m * DD + t] = (__bf16)(x - (float)h);
    red[t] = x * x;
    __syncthreads();
    for (int s = 64; s > 0; s >>= 1) {
        if (t < s) red[t] += red[t + s];
        __syncthreads();
    }
    if (t == 0) cn[m] = red[0];
    if (m == 0 && t == 0) { scal[0] = 0.0f; scal[1] = 0.0f; }
}

// ---------------------------------------------------------------------------
// Main RBF-matvec kernel. 256 threads = 8 waves; each wave owns 16 rows of A.
// MODE 0: A = X, writes preds, accumulates SSE into scal[0].
// MODE 1: A = centers, accumulates alpha^T (K_MM alpha) into scal[1].
// ---------------------------------------------------------------------------
template <int MODE>
__global__ __launch_bounds__(256) void flk_main(const float* __restrict__ Asrc, int nrows,
                                                const __bf16* __restrict__ Chi,
                                                const __bf16* __restrict__ Clo,
                                                const float* __restrict__ cn,
                                                const float* __restrict__ alpha,
                                                const float* __restrict__ Yv,
                                                const float* __restrict__ sigma_p,
                                                float* __restrict__ preds_out,
                                                float* __restrict__ scal, int M) {
    const int lane  = threadIdx.x & 31;
    const int wave  = threadIdx.x >> 5;
    const int r0    = lane & 15;        // row-in-tile (A) / col-in-chunk (B)
    const int kh    = lane >> 4;        // lane half selects K sub-range
    const int rowBase = blockIdx.x * 128 + wave * 16;
    const int row   = rowBase + r0;
    const int rowc  = row < nrows ? row : (nrows - 1);
    const float vm  = row < nrows ? 1.0f : 0.0f;

    const float sg  = sigma_p[0];
    const float sc  = LOG2E / (2.0f * sg * sg);   // exponent(base2) = 2*sc*dot - sc*xn - sc*cn
    const float two_sc = 2.0f * sc;

    // ---- Load & split A fragments (contiguous K layout matching B: lane holds
    //      K = kk+16*kh+(0..15)), compute ||x_row||^2.
    v16bf ah[4], al[4];
    float sumsq = 0.0f;
    #pragma unroll
    for (int ks = 0; ks < 4; ++ks) {
        const int kk = ks * 32;
        const float* p = Asrc + (size_t)rowc * DD + kk + 16 * kh;
        float4 f0 = *(const float4*)(p);
        float4 f1 = *(const float4*)(p + 4);
        float4 f2 = *(const float4*)(p + 8);
        float4 f3 = *(const float4*)(p + 12);
        float v[16] = { f0.x, f0.y, f0.z, f0.w, f1.x, f1.y, f1.z, f1.w,
                        f2.x, f2.y, f2.z, f2.w, f3.x, f3.y, f3.z, f3.w };
        #pragma unroll
        for (int e = 0; e < 16; ++e) {
            float x = v[e] * vm;                  // zero-pad invalid rows, no divergence
            sumsq = fmaf(x, x, sumsq);
            cvt_split(x, ah[ks], al[ks], e);
        }
    }
    const float xn = sumsq + __shfl_xor(sumsq, 16, 32);   // full row norm
    float u[8];
    #pragma unroll
    for (int r = 0; r < 8; ++r)                  // u[r] = sc * ||x_{r+8*kh}||^2
        u[r] = sc * __shfl(xn, r + 8 * kh, 32);

    v8f pred = {0.f, 0.f, 0.f, 0.f, 0.f, 0.f, 0.f, 0.f};

    // ---- Sweep all M centers in 16-column chunks.
    for (int jj = 0; jj < M; jj += 16) {
        const int col = jj + r0;
        v8f acc0 = {0.f, 0.f, 0.f, 0.f, 0.f, 0.f, 0.f, 0.f};
        v8f acc1 = {0.f, 0.f, 0.f, 0.f, 0.f, 0.f, 0.f, 0.f};
        #pragma unroll
        for (int ks = 0; ks < 4; ++ks) {
            const int kk = ks * 32;
            BU bh, bl;                            // B 32x16: lane = 16 contiguous K of one column
            const uint4* ph = (const uint4*)(Chi + (size_t)col * DD + kk + 16 * kh);
            const uint4* pl = (const uint4*)(Clo + (size_t)col * DD + kk + 16 * kh);
            bh.q[0] = ph[0]; bh.q[1] = ph[1];
            bl.q[0] = pl[0]; bl.q[1] = pl[1];
            acc0 = __builtin_amdgcn_wmma_f32_16x16x32_bf16(false, ah[ks], false, bh.v,
                                                           (short)0, acc0, false, false);
            acc1 = __builtin_amdgcn_wmma_f32_16x16x32_bf16(false, ah[ks], false, bl.v,
                                                           (short)0, acc1, false, false);
            acc1 = __builtin_amdgcn_wmma_f32_16x16x32_bf16(false, al[ks], false, bh.v,
                                                           (short)0, acc1, false, false);
        }
        const float t_j = sc * cn[col];
        const float a_j = alpha[col];
        #pragma unroll
        for (int r = 0; r < 8; ++r) {
            const float dot = acc0[r] + acc1[r];
            const float e2  = fmaf(dot, two_sc, -(u[r] + t_j));   // -||x-c||^2/(2s^2) * log2e
            pred[r] = fmaf(exp2f(e2), a_j, pred[r]);
        }
    }

    // ---- Reduce pred over the 16 lanes of each half (rows r+8*kh).
    #pragma unroll
    for (int r = 0; r < 8; ++r) {
        #pragma unroll
        for (int m = 1; m < 16; m <<= 1)
            pred[r] += __shfl_xor(pred[r], m, 32);
    }

    if (r0 == 0) {                                // lanes 0 and 16 write 8 rows each
        float local = 0.0f;
        #pragma unroll
        for (int r = 0; r < 8; ++r) {
            const int rr = rowBase + r + 8 * kh;
            if (rr < nrows) {
                const float p = pred[r];
                if (MODE == 0) {
                    preds_out[rr] = p;
                    const float d = p - Yv[rr];
                    local = fmaf(d, d, local);
                } else {
                    local = fmaf(alpha[rr], p, local);
                }
            }
        }
        atomicAdd(&scal[MODE], local);
    }
}

__global__ void flk_finalize(const float* __restrict__ scal,
                             const float* __restrict__ penalty,
                             float* __restrict__ out, int N) {
    out[0] = scal[0] / (float)N + __expf(-penalty[0]) * scal[1];
}

// ---------------------------------------------------------------------------
extern "C" void kernel_launch(void* const* d_in, const int* in_sizes, int n_in,
                              void* d_out, int out_size, void* d_ws, size_t ws_size,
                              hipStream_t stream) {
    const float* X       = (const float*)d_in[0];
    const float* Y       = (const float*)d_in[1];
    const float* centers = (const float*)d_in[2];
    const float* alpha   = (const float*)d_in[3];
    const float* sigma   = (const float*)d_in[4];
    const float* penalty = (const float*)d_in[5];
    float* out = (float*)d_out;

    const int N = in_sizes[0] / DD;      // 100000
    const int M = in_sizes[2] / DD;      // 2048

    // Workspace layout: Chi[M*128] bf16, Clo[M*128] bf16, cn[M] f32, scal[2] f32
    __bf16* Chi = (__bf16*)d_ws;
    __bf16* Clo = Chi + (size_t)M * DD;
    float*  cn  = (float*)(Clo + (size_t)M * DD);
    float*  scal = cn + M;

    flk_prep<<<M, 128, 0, stream>>>(centers, Chi, Clo, cn, scal);

    const int gridN = (N + 127) / 128;
    flk_main<0><<<gridN, 256, 0, stream>>>(X, N, Chi, Clo, cn, alpha, Y, sigma,
                                           out + 1, scal, M);
    const int gridM = (M + 127) / 128;
    flk_main<1><<<gridM, 256, 0, stream>>>(centers, M, Chi, Clo, cn, alpha, nullptr, sigma,
                                           nullptr, scal, M);

    flk_finalize<<<1, 1, 0, stream>>>(scal, penalty, out, N);
}
